// MetaPathGNN_2405181686102
// MI455X (gfx1250) — matrix operand order repaired and checked by reading, not compiled, not run on hardware
//
#include <hip/hip_runtime.h>
#include <hip/hip_bf16.h>

#define NU 20000
#define NI 40000
#define NF 5000
#define N_ALL 80000
#define N_UI 60000
#define E_META 600000
#define E_UI 600000
#define HEADS 4
#define HD 64
#define OUTD 256   // HEADS*HD
#define SLOPE 0.2f

typedef __attribute__((ext_vector_type(16))) _Float16 v16h;
typedef __attribute__((ext_vector_type(8)))  float    v8f;

// ---------------------------------------------------------------------------
// WMMA GEMM: C[M x N] = A[M x K] @ W[K x N] (+ bias row, optional).
// W is pre-converted to f16 -> B fragments are per-element loads, stride N over K
// (zero cvt). grid = (ceil(M/64), N/64),
// block = 128 threads = 4 waves; wave owns a 16x64 strip: A fragment once
// per K-step, 4 back-to-back WMMAs on independent accumulators.
// TA = float (converted on load) or _Float16 (direct b128 loads).
// TC = float or _Float16 (down-converted on store).
// K multiple of 32 (here 64/128/256), N multiple of 64.
// Fragments per CDNA5 ISA 7.12.2 (wave32):
//   A 16x32 f16 : lane<16 holds row=lane, elems 0..7 = K 0..7, 8..15 = K16..23
//                 lane>=16 holds row=lane-16, elems 0..7 = K 8..15, 8..15 = K24..31
//   B 32x16 f16 : lane holds N=lane&15, elems = same K order as A
//   C/D 16x16 f32: vgpr r -> row r + 8*(lane>=16), col = lane&15
// ---------------------------------------------------------------------------
template <typename TA, typename TC>
__global__ void wmma_gemm_f16(const TA* __restrict__ A,
                              const _Float16* __restrict__ W,
                              const float* __restrict__ bias,
                              TC* __restrict__ C,
                              int M, int K, int N) {
  const int wave = threadIdx.x >> 5;
  const int lane = threadIdx.x & 31;
  const int row0 = (blockIdx.x * 4 + wave) * 16;
  if (row0 >= M) return;  // wave-uniform exit; live waves keep EXEC all-1s
  const int col0 = blockIdx.y * 64;
  const int arow = row0 + (lane & 15);
  const int koff = (lane >> 4) * 8;

  v8f acc[4];
#pragma unroll
  for (int t = 0; t < 4; ++t)
#pragma unroll
    for (int r = 0; r < 8; ++r) acc[t][r] = 0.0f;

  for (int kk = 0; kk < K; kk += 32) {
    // ---- A fragment (shared across the 4 N-tiles) ----
    v16h a;
    if (arow < M) {
      const TA* ap = A + (size_t)arow * K + kk + koff;
#pragma unroll
      for (int j = 0; j < 8; ++j) {
        a[j]     = (_Float16)ap[j];
        a[8 + j] = (_Float16)ap[16 + j];
      }
    } else {
#pragma unroll
      for (int j = 0; j < 16; ++j) a[j] = (_Float16)0.0f;
    }

    // ---- 4 B fragments: column n = lane&15, elements strided over K (table order) ----
    const _Float16* wrow = W + (size_t)kk * N + col0 + (lane & 15);
    if (kk + 32 < K) __builtin_prefetch(wrow + (size_t)32 * N, 0, 1);
    v16h b[4];
#pragma unroll
    for (int t = 0; t < 4; ++t)
#pragma unroll
      for (int j = 0; j < 8; ++j) {
        b[t][j]     = wrow[(size_t)(koff + j) * N + t * 16];
        b[t][8 + j] = wrow[(size_t)(16 + koff + j) * N + t * 16];
      }

    // ---- 4 WMMAs, independent accumulators ----
#pragma unroll
    for (int t = 0; t < 4; ++t) {
      acc[t] = __builtin_amdgcn_wmma_f32_16x16x32_f16(
          /*neg_a=*/false, a, /*neg_b=*/false, b[t],
          /*c_mod=*/(short)0, acc[t], /*reuse_a=*/false, /*reuse_b=*/false);
    }
  }

  const int rbase = (lane >> 4) * 8;
  const int cbase = col0 + (lane & 15);
#pragma unroll
  for (int t = 0; t < 4; ++t) {
    const int col = cbase + t * 16;
    const float bv = bias ? bias[col] : 0.0f;
#pragma unroll
    for (int r = 0; r < 8; ++r) {
      const int row = row0 + rbase + r;
      if (row < M) C[(size_t)row * N + col] = (TC)(acc[t][r] + bv);
    }
  }
}

// f32 -> f16 weight down-conversion (run once per launch; ~200K elems total)
__global__ void convert_f16(const float* __restrict__ src,
                            _Float16* __restrict__ dst, int n) {
  int i = blockIdx.x * blockDim.x + threadIdx.x;
  if (i < n) dst[i] = (_Float16)src[i];
}

// ---------------------------------------------------------------------------
// Attention scores: a_s[n,h] = dot(h[n,h,:], a_src[h,:]); same for a_d.
// ---------------------------------------------------------------------------
__global__ void attn_scores(const float* __restrict__ hbuf,
                            const float* __restrict__ a_src,
                            const float* __restrict__ a_dst,
                            float* __restrict__ as_, float* __restrict__ ad_,
                            int Nn) {
  int i = blockIdx.x * blockDim.x + threadIdx.x;  // node*HEADS + head
  if (i >= Nn * HEADS) return;
  int node = i >> 2, head = i & 3;
  const float* hp = hbuf + (size_t)node * OUTD + head * HD;
  const float* sp = a_src + head * HD;
  const float* dp = a_dst + head * HD;
  float vs = 0.0f, vd = 0.0f;
#pragma unroll 4
  for (int k = 0; k < HD; ++k) { float x = hp[k]; vs += x * sp[k]; vd += x * dp[k]; }
  as_[i] = vs;
  ad_[i] = vd;
}

__global__ void init_stats(float* __restrict__ m_, float* __restrict__ den, int n) {
  int i = blockIdx.x * blockDim.x + threadIdx.x;
  if (i >= n) return;
  m_[i] = -__FLT_MAX__;
  den[i] = 0.0f;
}

__global__ void init_out_bias(float* __restrict__ out, const float* __restrict__ bias,
                              int Nn) {
  size_t i = (size_t)blockIdx.x * blockDim.x + threadIdx.x;
  if (i >= (size_t)Nn * OUTD) return;
  out[i] = bias[i % OUTD];
}

__device__ __forceinline__ void atomicMaxFloat(float* addr, float val) {
  if (val >= 0.0f)
    atomicMax((int*)addr, __float_as_int(val));
  else
    atomicMin((unsigned int*)addr, __float_as_uint(val));
}

__device__ __forceinline__ float leaky(float x) { return x > 0.0f ? x : SLOPE * x; }

// segment max over destinations (incl. implicit self loops for e >= E)
__global__ void edge_max(const int* __restrict__ e_src, const int* __restrict__ e_dst,
                         int E, int Nn,
                         const float* __restrict__ as_, const float* __restrict__ ad_,
                         float* __restrict__ m_) {
  int i = blockIdx.x * blockDim.x + threadIdx.x;  // edge*HEADS + head
  if (i >= (E + Nn) * HEADS) return;
  int e = i >> 2, hd = i & 3;
  int s, d;
  if (e < E) { s = e_src[e]; d = e_dst[e]; } else { s = d = e - E; }
  float x = leaky(as_[s * HEADS + hd] + ad_[d * HEADS + hd]);
  atomicMaxFloat(&m_[d * HEADS + hd], x);
}

// denom = segment_sum(exp(e - m[dst]))
__global__ void edge_expsum(const int* __restrict__ e_src, const int* __restrict__ e_dst,
                            int E, int Nn,
                            const float* __restrict__ as_, const float* __restrict__ ad_,
                            const float* __restrict__ m_, float* __restrict__ den) {
  int i = blockIdx.x * blockDim.x + threadIdx.x;
  if (i >= (E + Nn) * HEADS) return;
  int e = i >> 2, hd = i & 3;
  int s, d;
  if (e < E) { s = e_src[e]; d = e_dst[e]; } else { s = d = e - E; }
  float x = leaky(as_[s * HEADS + hd] + ad_[d * HEADS + hd]);
  float p = __expf(x - m_[d * HEADS + hd]);
  atomicAdd(&den[d * HEADS + hd], p);
}

// out[dst] += h[src] * alpha ; one wave per edge, 256 channels per edge.
__global__ void edge_aggr(const int* __restrict__ e_src, const int* __restrict__ e_dst,
                          int E, int Nn,
                          const float* __restrict__ as_, const float* __restrict__ ad_,
                          const float* __restrict__ m_, const float* __restrict__ den,
                          const float* __restrict__ hbuf, float* __restrict__ out) {
  int wave = threadIdx.x >> 5, lane = threadIdx.x & 31;
  int e = blockIdx.x * 8 + wave;
  if (e >= E + Nn) return;
  int s, d;
  if (e < E) { s = e_src[e]; d = e_dst[e]; } else { s = d = e - E; }
  float alpha[HEADS];
#pragma unroll
  for (int hd = 0; hd < HEADS; ++hd) {
    float x = leaky(as_[s * HEADS + hd] + ad_[d * HEADS + hd]);
    float p = __expf(x - m_[d * HEADS + hd]);
    alpha[hd] = p / (den[d * HEADS + hd] + 1e-16f);
  }
  const float* hs = hbuf + (size_t)s * OUTD;
  float* od = out + (size_t)d * OUTD;
#pragma unroll
  for (int it = 0; it < 8; ++it) {
    int c = lane + 32 * it;
    atomicAdd(&od[c], hs[c] * alpha[c >> 6]);
  }
}

// combined[r] = uqu[r] (r < NU) else iqi[r], stored as f16 (feeds stage-2 GEMM A)
__global__ void build_combined(const float* __restrict__ out1,
                               const float* __restrict__ out2,
                               _Float16* __restrict__ comb) {
  size_t i = (size_t)blockIdx.x * blockDim.x + threadIdx.x;
  if (i >= (size_t)N_UI * OUTD) return;
  size_t row = i / OUTD;
  comb[i] = (_Float16)((row < NU) ? out1[i] : out2[i]);
}

__global__ void copy_f32(const float* __restrict__ src, float* __restrict__ dst,
                         size_t n) {
  size_t i = (size_t)blockIdx.x * blockDim.x + threadIdx.x;
  if (i < n) dst[i] = src[i];
}

// ---------------------------------------------------------------------------
// Host-side GAT driver (all launches on `stream`); x and W are f16.
// ---------------------------------------------------------------------------
static void run_gat(const _Float16* x, int Nn, int fin,
                    const _Float16* W, const float* a_src, const float* a_dst,
                    const float* bias, const int* edges, int E,
                    float* hbuf, float* out,
                    float* as_, float* ad_, float* m_, float* den,
                    hipStream_t stream) {
  dim3 gg((Nn + 63) / 64, OUTD / 64);
  wmma_gemm_f16<_Float16, float><<<gg, 128, 0, stream>>>(x, W, nullptr, hbuf,
                                                         Nn, fin, OUTD);

  int t = Nn * HEADS;
  attn_scores<<<(t + 255) / 256, 256, 0, stream>>>(hbuf, a_src, a_dst, as_, ad_, Nn);
  init_stats<<<(t + 255) / 256, 256, 0, stream>>>(m_, den, t);
  size_t no = (size_t)Nn * OUTD;
  init_out_bias<<<(unsigned)((no + 255) / 256), 256, 0, stream>>>(out, bias, Nn);

  const int* e_src = edges;
  const int* e_dst = edges + E;
  int tot4 = (E + Nn) * HEADS;
  edge_max<<<(tot4 + 255) / 256, 256, 0, stream>>>(e_src, e_dst, E, Nn, as_, ad_, m_);
  edge_expsum<<<(tot4 + 255) / 256, 256, 0, stream>>>(e_src, e_dst, E, Nn, as_, ad_, m_, den);
  int totE = E + Nn;
  edge_aggr<<<(totE + 7) / 8, 256, 0, stream>>>(e_src, e_dst, E, Nn, as_, ad_, m_, den,
                                                hbuf, out);
}

extern "C" void kernel_launch(void* const* d_in, const int* in_sizes, int n_in,
                              void* d_out, int out_size, void* d_ws, size_t ws_size,
                              hipStream_t stream) {
  const float* user_x = (const float*)d_in[0];
  const float* item_x = (const float*)d_in[1];
  const float* uf0    = (const float*)d_in[2];
  const float* uf1    = (const float*)d_in[3];
  const float* if0    = (const float*)d_in[4];
  const float* if1    = (const float*)d_in[5];
  const float* Wu  = (const float*)d_in[6];
  const float* bu  = (const float*)d_in[7];
  const float* Wi  = (const float*)d_in[8];
  const float* bi  = (const float*)d_in[9];
  const float* Wuf = (const float*)d_in[10];
  const float* buf = (const float*)d_in[11];
  const float* Wif = (const float*)d_in[12];
  const float* bif = (const float*)d_in[13];
  const float* g1_W  = (const float*)d_in[14];
  const float* g1_as = (const float*)d_in[15];
  const float* g1_ad = (const float*)d_in[16];
  const float* g1_b  = (const float*)d_in[17];
  const float* g2_W  = (const float*)d_in[18];
  const float* g2_as = (const float*)d_in[19];
  const float* g2_ad = (const float*)d_in[20];
  const float* g2_b  = (const float*)d_in[21];
  const float* g3_W  = (const float*)d_in[22];
  const float* g3_as = (const float*)d_in[23];
  const float* g3_ad = (const float*)d_in[24];
  const float* g3_b  = (const float*)d_in[25];
  const float* g4_W  = (const float*)d_in[26];
  const float* g4_as = (const float*)d_in[27];
  const float* g4_ad = (const float*)d_in[28];
  const float* g4_b  = (const float*)d_in[29];
  const int* edges_uqu = (const int*)d_in[30];
  const int* edges_iqi = (const int*)d_in[31];
  const int* edges_ui  = (const int*)d_in[32];
  const int* edges_iu  = (const int*)d_in[33];
  float* out = (float*)d_out;

  // Workspace layout: f32 buffers first, then f16 activations + f16 weights.
  float* ws   = (float*)d_ws;
  float* hbuf = ws;                                   // N_ALL * 256  f32
  float* out1 = hbuf + (size_t)N_ALL * OUTD;          // N_ALL * 256  f32
  float* out2 = out1 + (size_t)N_ALL * OUTD;          // N_ALL * 256  f32
  float* as_  = out2 + (size_t)N_ALL * OUTD;          // N_ALL * 4    f32
  float* ad_  = as_  + (size_t)N_ALL * HEADS;
  float* m_   = ad_  + (size_t)N_ALL * HEADS;
  float* den  = m_   + (size_t)N_ALL * HEADS;
  _Float16* emb  = (_Float16*)(den + (size_t)N_ALL * HEADS);  // N_ALL * 64  f16
  _Float16* comb = emb + (size_t)N_ALL * HD;                  // N_UI * 256  f16
  _Float16* wWu  = comb + (size_t)N_UI * OUTD;  // 128*64
  _Float16* wWi  = wWu  + 128 * HD;             // 128*64
  _Float16* wWuf = wWi  + 128 * HD;             // 64*64
  _Float16* wWif = wWuf + HD * HD;              // 64*64
  _Float16* wG1  = wWif + HD * HD;              // 64*256
  _Float16* wG2  = wG1  + HD * OUTD;            // 64*256
  _Float16* wG3  = wG2  + HD * OUTD;            // 256*256
  _Float16* wG4  = wG3  + OUTD * OUTD;          // 256*256

  // ---- weight down-conversion (per launch; deterministic, no cached state) ----
  convert_f16<<<(128 * HD + 255) / 256, 256, 0, stream>>>(Wu, wWu, 128 * HD);
  convert_f16<<<(128 * HD + 255) / 256, 256, 0, stream>>>(Wi, wWi, 128 * HD);
  convert_f16<<<(HD * HD + 255) / 256, 256, 0, stream>>>(Wuf, wWuf, HD * HD);
  convert_f16<<<(HD * HD + 255) / 256, 256, 0, stream>>>(Wif, wWif, HD * HD);
  convert_f16<<<(HD * OUTD + 255) / 256, 256, 0, stream>>>(g1_W, wG1, HD * OUTD);
  convert_f16<<<(HD * OUTD + 255) / 256, 256, 0, stream>>>(g2_W, wG2, HD * OUTD);
  convert_f16<<<(OUTD * OUTD + 255) / 256, 256, 0, stream>>>(g3_W, wG3, OUTD * OUTD);
  convert_f16<<<(OUTD * OUTD + 255) / 256, 256, 0, stream>>>(g4_W, wG4, OUTD * OUTD);

  // ---- Stage 0: projections into all_emb (N_ALL x 64, f16) ----
  wmma_gemm_f16<float, _Float16><<<dim3((NU + 63) / 64, 1), 128, 0, stream>>>(
      user_x, wWu, bu, emb, NU, 128, HD);
  wmma_gemm_f16<float, _Float16><<<dim3((NI + 63) / 64, 1), 128, 0, stream>>>(
      item_x, wWi, bi, emb + (size_t)NU * HD, NI, 128, HD);
  _Float16* fbase = emb + (size_t)(NU + NI) * HD;
  wmma_gemm_f16<float, _Float16><<<dim3((NF + 63) / 64, 1), 128, 0, stream>>>(
      uf0, wWuf, buf, fbase, NF, HD, HD);
  wmma_gemm_f16<float, _Float16><<<dim3((NF + 63) / 64, 1), 128, 0, stream>>>(
      uf1, wWuf, buf, fbase + (size_t)NF * HD, NF, HD, HD);
  wmma_gemm_f16<float, _Float16><<<dim3((NF + 63) / 64, 1), 128, 0, stream>>>(
      if0, wWif, bif, fbase + (size_t)2 * NF * HD, NF, HD, HD);
  wmma_gemm_f16<float, _Float16><<<dim3((NF + 63) / 64, 1), 128, 0, stream>>>(
      if1, wWif, bif, fbase + (size_t)3 * NF * HD, NF, HD, HD);

  // ---- Stage 1: meta-path GATs on all_emb ----
  run_gat(emb, N_ALL, HD, wG1, g1_as, g1_ad, g1_b, edges_uqu, E_META,
          hbuf, out1, as_, ad_, m_, den, stream);   // uqu
  run_gat(emb, N_ALL, HD, wG2, g2_as, g2_ad, g2_b, edges_iqi, E_META,
          hbuf, out2, as_, ad_, m_, den, stream);   // iqi

  // ---- combined = concat(uqu[:NU], iqi[NU:NU+NI]) as f16 ----
  size_t nc = (size_t)N_UI * OUTD;
  build_combined<<<(unsigned)((nc + 255) / 256), 256, 0, stream>>>(out1, out2, comb);

  // ---- Stage 2: user/item GATs on combined (reuse out1/out2) ----
  run_gat(comb, N_UI, OUTD, wG4, g4_as, g4_ad, g4_b, edges_ui, E_UI,
          hbuf, out1, as_, ad_, m_, den, stream);   // -> H_hat_u = out1[:NU]
  size_t nu = (size_t)NU * OUTD;
  copy_f32<<<(unsigned)((nu + 255) / 256), 256, 0, stream>>>(out1, out, nu);

  run_gat(comb, N_UI, OUTD, wG3, g3_as, g3_ad, g3_b, edges_iu, E_UI,
          hbuf, out2, as_, ad_, m_, den, stream);   // -> H_hat_i = out2[NU:NU+NI]
  size_t ni = (size_t)NI * OUTD;
  copy_f32<<<(unsigned)((ni + 255) / 256), 256, 0, stream>>>(out2 + nu, out + nu, ni);
}
